// EdgeMessageBlock_16535624090328
// MI455X (gfx1250) — matrix operand order repaired and checked by reading, not compiled, not run on hardware
//
#include <hip/hip_runtime.h>
#include <hip/hip_bf16.h>

typedef _Float16 v16h __attribute__((ext_vector_type(16)));
typedef _Float16 v8h  __attribute__((ext_vector_type(8)));
typedef float    v8f  __attribute__((ext_vector_type(8)));
typedef float    v4f  __attribute__((ext_vector_type(4)));

#define WMMA_F16(A,B,C) __builtin_amdgcn_wmma_f32_16x16x32_f16(false,(A),false,(B),(short)0,(C),false,false)

// Packed f16 weight layout in d_ws (element counts in halves):
//  WT1 (transposed W1, padded): 64 rows(n) x 168 cols(k, 131 valid)  @ 0
//  WT2: 64 x 72 (64 valid)                                           @ 10752
//  WT3: 64 x 72                                                      @ 15360
#define W1_STR 168
#define W2_OFF 10752
#define W2_STR 72
#define W3_OFF 15360
#define W3_STR 72
#define W_TOT  19968   // halves (= 39936 bytes)
#define X_STR  72      // per-row stride of the per-wave activation tile (halves)

__global__ __launch_bounds__(256) void prep_weights_kernel(
    const float* __restrict__ W1, const float* __restrict__ W2,
    const float* __restrict__ W3, _Float16* __restrict__ wp)
{
  const int tid = threadIdx.x;
  for (int i = tid; i < 64 * W1_STR; i += 256) {
    int n = i / W1_STR, k = i % W1_STR;
    wp[i] = (k < 131) ? (_Float16)W1[k * 64 + n] : (_Float16)0.0f;
  }
  for (int i = tid; i < 64 * W2_STR; i += 256) {
    int n = i / W2_STR, k = i % W2_STR;
    _Float16 z = (_Float16)0.0f;
    wp[W2_OFF + i] = (k < 64) ? (_Float16)W2[k * 64 + n] : z;
    wp[W3_OFF + i] = (k < 64) ? (_Float16)W3[k * 64 + n] : z;
  }
}

__global__ void zero_kernel(float* __restrict__ p, int n) {
  int i = blockIdx.x * blockDim.x + threadIdx.x;
  if (i < n) p[i] = 0.0f;
}

__device__ __forceinline__ v8f vzero8() {
  v8f r;
  #pragma unroll
  for (int i = 0; i < 8; ++i) r[i] = 0.0f;
  return r;
}
__device__ __forceinline__ v16h hzero16() {
  v16h r;
  #pragma unroll
  for (int i = 0; i < 16; ++i) r[i] = (_Float16)0.0f;
  return r;
}
__device__ __forceinline__ v8h cvt8(v4f a, v4f b) {
  v8h r;
  r[0]=(_Float16)a[0]; r[1]=(_Float16)a[1]; r[2]=(_Float16)a[2]; r[3]=(_Float16)a[3];
  r[4]=(_Float16)b[0]; r[5]=(_Float16)b[1]; r[6]=(_Float16)b[2]; r[7]=(_Float16)b[3];
  return r;
}
__device__ __forceinline__ v16h cat16(v8h lo, v8h hi) {
  return __builtin_shufflevector(lo, hi, 0,1,2,3,4,5,6,7,8,9,10,11,12,13,14,15);
}

__global__ __launch_bounds__(256) void edge_mlp_wmma_kernel(
    const float* __restrict__ h, const float* __restrict__ theta,
    const int* __restrict__ ei, const float* __restrict__ Kn,
    const float* __restrict__ alive,
    const float* __restrict__ b1, const float* __restrict__ b2,
    const float* __restrict__ b3,
    const _Float16* __restrict__ wp, float* __restrict__ out, int E)
{
  __shared__ __align__(16) _Float16 lds_w[W_TOT];
  __shared__ __align__(16) _Float16 lds_x[8 * 16 * X_STR];
  __shared__ float lds_scale[8 * 16];
  __shared__ int   lds_dst[8 * 16];

  const int tid = threadIdx.x;
  { // cooperative stage of packed f16 weights into LDS (16B chunks)
    const uint4* s = (const uint4*)wp;
    uint4* d = (uint4*)lds_w;
    #pragma unroll 1
    for (int i = tid; i < W_TOT / 8; i += 256) d[i] = s[i];
  }
  __syncthreads();

  const int wave = tid >> 5, lane = tid & 31;
  const int hl = lane & 15, hi = lane >> 4;
  const int e0 = (blockIdx.x * 8 + wave) * 16;
  if (e0 >= E) return;

  _Float16* xw  = lds_x    + wave * 16 * X_STR;
  float*    scw = lds_scale + wave * 16;
  int*      dsw = lds_dst   + wave * 16;

  // --- per-edge-row scalars (row M = hl; both lane halves compute them) ---
  int erow = e0 + hl;
  const float valid = (erow < E) ? 1.0f : 0.0f;
  if (erow >= E) erow = E - 1;
  const int si = ei[erow];
  const int di = ei[E + erow];
  if (hi == 0) {
    scw[hl] = alive[si] * alive[di] * valid;
    dsw[hl] = di;
  }
  const float delta = theta[si] - theta[di];
  const float sd = sinf(delta), cd = cosf(delta);
  const float ke = Kn[di];

  const float* hs = h + (long)si * 64;
  const float* hd = h + (long)di * 64;

  // A-fragment (16x32 f16): lane half hi holds K = off+hi*8..+7 and off+16+hi*8..+7
  auto afrag = [&](const float* hb, int off) -> v16h {
    const int k0 = off + hi * 8;
    v4f a0 = *(const v4f*)(hb + k0);
    v4f a1 = *(const v4f*)(hb + k0 + 4);
    v4f a2 = *(const v4f*)(hb + k0 + 16);
    v4f a3 = *(const v4f*)(hb + k0 + 20);
    return cat16(cvt8(a0, a1), cvt8(a2, a3));
  };
  // B-fragment (32x16 f16) from LDS weights: lane half hi holds K = kc*32+hi*8..+7 and kc*32+16+hi*8..+7
  // of output column n = nt*16+hl (weights stored transposed, row n contiguous in k)
  auto bfrag = [&](const _Float16* wb, int stride, int nt, int kc) -> v16h {
    const _Float16* p = wb + (nt * 16 + hl) * stride + kc * 32 + hi * 8;
    return cat16(*(const v8h*)p, *(const v8h*)(p + 16));
  };

  const _Float16* w1 = lds_w;
  const _Float16* w2 = lds_w + W2_OFF;
  const _Float16* w3 = lds_w + W3_OFF;

  v8f acc[4];
  #pragma unroll
  for (int nt = 0; nt < 4; ++nt) acc[nt] = vzero8();

  // ---------------- layer 1: [16 x 160] @ [160 x 64] ----------------
  #pragma unroll
  for (int kc = 0; kc < 4; ++kc) {
    const float* hb = (kc < 2) ? hd : hs;   // k 0..63 = h_dst, 64..127 = h_src
    v16h a = afrag(hb, (kc & 1) * 32);
    #pragma unroll
    for (int nt = 0; nt < 4; ++nt)
      acc[nt] = WMMA_F16(a, bfrag(w1, W1_STR, nt, kc), acc[nt]);
  }
  { // kc = 4 : features 128..130 = sin, cos, K_edge; rest zero-padded
    v16h a = hzero16();
    const _Float16 z = (_Float16)0.0f;
    a[0] = (hi == 0) ? (_Float16)sd : z;
    a[1] = (hi == 0) ? (_Float16)cd : z;
    a[2] = (hi == 0) ? (_Float16)ke : z;
    #pragma unroll
    for (int nt = 0; nt < 4; ++nt)
      acc[nt] = WMMA_F16(a, bfrag(w1, W1_STR, nt, 4), acc[nt]);
  }

  // bias + ReLU + stash activations (f16) into the per-wave LDS tile.
  // D layout: vgpr r of lane = element (M = r + 8*hi, N = nt*16 + hl).
  auto act_store = [&](const float* bias) {
    #pragma unroll
    for (int nt = 0; nt < 4; ++nt) {
      const float bv = bias[nt * 16 + hl];
      #pragma unroll
      for (int r = 0; r < 8; ++r) {
        float v = acc[nt][r] + bv;
        v = fmaxf(v, 0.0f);
        xw[(r + 8 * hi) * X_STR + nt * 16 + hl] = (_Float16)v;
      }
    }
  };
  // reload activation tile as two A-fragments (K = 0..31, 32..63)
  auto xload = [&](v16h& xa0, v16h& xa1) {
    const _Float16* p = xw + hl * X_STR;
    xa0 = cat16(*(const v8h*)(p + hi * 8),      *(const v8h*)(p + 16 + hi * 8));
    xa1 = cat16(*(const v8h*)(p + 32 + hi * 8), *(const v8h*)(p + 48 + hi * 8));
  };

  v16h xa0, xa1;

  // ---------------- layer 2 ----------------
  act_store(b1);
  __builtin_amdgcn_sched_barrier(0);   // keep ds_store -> ds_load order
  xload(xa0, xa1);
  __builtin_amdgcn_sched_barrier(0);
  #pragma unroll
  for (int nt = 0; nt < 4; ++nt) acc[nt] = vzero8();
  #pragma unroll
  for (int nt = 0; nt < 4; ++nt) {
    acc[nt] = WMMA_F16(xa0, bfrag(w2, W2_STR, nt, 0), acc[nt]);
    acc[nt] = WMMA_F16(xa1, bfrag(w2, W2_STR, nt, 1), acc[nt]);
  }

  // ---------------- layer 3 ----------------
  act_store(b2);
  __builtin_amdgcn_sched_barrier(0);
  xload(xa0, xa1);
  __builtin_amdgcn_sched_barrier(0);
  #pragma unroll
  for (int nt = 0; nt < 4; ++nt) acc[nt] = vzero8();
  #pragma unroll
  for (int nt = 0; nt < 4; ++nt) {
    acc[nt] = WMMA_F16(xa0, bfrag(w3, W3_STR, nt, 0), acc[nt]);
    acc[nt] = WMMA_F16(xa1, bfrag(w3, W3_STR, nt, 1), acc[nt]);
  }

  // ---------------- epilogue: bias3, alive scale, scatter-add ----------------
  float rs[8]; int rd[8];
  #pragma unroll
  for (int r = 0; r < 8; ++r) { rs[r] = scw[r + 8 * hi]; rd[r] = dsw[r + 8 * hi]; }
  #pragma unroll
  for (int nt = 0; nt < 4; ++nt) {
    const float bv = b3[nt * 16 + hl];
    #pragma unroll
    for (int r = 0; r < 8; ++r) {
      const float v = (acc[nt][r] + bv) * rs[r];
      unsafeAtomicAdd(out + (long)rd[r] * 64 + nt * 16 + hl, v);
    }
  }
}

extern "C" void kernel_launch(void* const* d_in, const int* in_sizes, int n_in,
                              void* d_out, int out_size, void* d_ws, size_t ws_size,
                              hipStream_t stream) {
  const float* h     = (const float*)d_in[0];
  const float* theta = (const float*)d_in[1];
  const int*   ei    = (const int*)d_in[2];
  const float* Kn    = (const float*)d_in[3];
  const float* alive = (const float*)d_in[4];
  const float* W1 = (const float*)d_in[5];  const float* b1 = (const float*)d_in[6];
  const float* W2 = (const float*)d_in[7];  const float* b2 = (const float*)d_in[8];
  const float* W3 = (const float*)d_in[9];  const float* b3 = (const float*)d_in[10];

  const int E = in_sizes[2] / 2;            // edge_index is [2, E]
  float* out = (float*)d_out;
  _Float16* wp = (_Float16*)d_ws;           // 39936 bytes used

  prep_weights_kernel<<<1, 256, 0, stream>>>(W1, W2, W3, wp);
  zero_kernel<<<(out_size + 255) / 256, 256, 0, stream>>>(out, out_size);

  const int tiles  = (E + 15) / 16;         // one wave per 16-edge tile
  const int blocks = (tiles + 7) / 8;       // 8 waves (256 threads) per block
  edge_mlp_wmma_kernel<<<blocks, 256, 0, stream>>>(
      h, theta, ei, Kn, alive, b1, b2, b3, wp, out, E);
}
